// WaveEncoder_46359876993378
// MI455X (gfx1250) — matrix operand order repaired and checked by reading, not compiled, not run on hardware
//
#include <hip/hip_runtime.h>

typedef __attribute__((ext_vector_type(16))) _Float16 v16h;
typedef __attribute__((ext_vector_type(8)))  float    v8f;

#define S_LEN   4096
#define DMODEL  1024
#define NFREQ   64
#define NBATCH  4

// Fused WaveEncoder:
//   coef_real = cos(phase) @ A, coef_imag = sin(phase) @ A   (WMMA, in-register)
//   out_real[b,s,d] = coef_real[s,d] * emb[tok[b,s], d]
//   out_imag[b,s,d] = coef_imag[s,d] * emb[tok[b,s], d]
// Grid: (DMODEL/128, S_LEN/16); block: 256 threads = 8 waves; each wave owns a
// 16(s) x 16(d) tile, two K=32 WMMA steps cover F=64.
// Outputs are streamed with non-temporal stores (TH=NT) so the 128 MiB
// write-once stream does not evict the L2-resident embedding/amplitude data.
__global__ __launch_bounds__(256)
void wave_encoder_fused(const int* __restrict__ tokens,
                        const int* __restrict__ t_ptr,
                        const float* __restrict__ emb,
                        const float* __restrict__ amp,
                        const float* __restrict__ freq,
                        float* __restrict__ out)
{
    const int wave = threadIdx.x >> 5;            // 0..7
    const int lane = threadIdx.x & 31;
    const int half = lane >> 4;                   // 0 or 1
    const int ml   = lane & 15;

    const int s0 = blockIdx.y << 4;               // 16 s-rows per block
    const int d0 = (blockIdx.x << 7) + (wave << 4); // 16 d-cols per wave
    const int dcol = d0 + ml;

    const float t = (float)(*t_ptr);
    const float TWO_PI = 6.2831853071795864769f;

    // ---------------- A operands: cos/sin(phase), 16x32 f16, ISA layout ----
    // lane<16: row M=ml, halfs[0..7]=K(kbase..kbase+7), halfs[8..15]=K(+16)
    // lane>=16: same rows, K shifted by 8.
    const float ks = TWO_PI * (float)(s0 + ml) * (1.0f / (float)S_LEN);
    v16h a_cos0, a_cos1, a_sin0, a_sin1;
    {
        const int kb0 = half * 8;        // chunk 0: K in [0,32)
        const int kb1 = 32 + half * 8;   // chunk 1: K in [32,64)
        #pragma unroll
        for (int j = 0; j < 8; ++j) {
            float p00 = ks - TWO_PI * freq[kb0 + j]      * t;
            float p01 = ks - TWO_PI * freq[kb0 + 16 + j] * t;
            float p10 = ks - TWO_PI * freq[kb1 + j]      * t;
            float p11 = ks - TWO_PI * freq[kb1 + 16 + j] * t;
            a_cos0[j]     = (_Float16)__cosf(p00);
            a_cos0[8 + j] = (_Float16)__cosf(p01);
            a_cos1[j]     = (_Float16)__cosf(p10);
            a_cos1[8 + j] = (_Float16)__cosf(p11);
            a_sin0[j]     = (_Float16)__sinf(p00);
            a_sin0[8 + j] = (_Float16)__sinf(p01);
            a_sin1[j]     = (_Float16)__sinf(p10);
            a_sin1[8 + j] = (_Float16)__sinf(p11);
        }
    }

    // ---------------- B operands: amplitude columns, 32x16 f16 -------------
    // lane holds column N=ml; halfs[0..7]=K(kb+8*half..+7), halfs[8..15]=K(kb+16+8*half..+7)
    v16h b0, b1;
    {
        const int kb0 = half * 8;
        const int kb1 = 32 + half * 8;
        #pragma unroll
        for (int j = 0; j < 16; ++j) {
            b0[j] = (_Float16)amp[(kb0 + j + (j & 8)) * DMODEL + dcol];
            b1[j] = (_Float16)amp[(kb1 + j + (j & 8)) * DMODEL + dcol];
        }
    }

    // ---------------- WMMA: chain K=0..31 then K=32..63 --------------------
    v8f c_cos = {};
    v8f c_sin = {};
    c_cos = __builtin_amdgcn_wmma_f32_16x16x32_f16(false, a_cos0, false, b0,
                                                   (short)0, c_cos, false, false);
    c_cos = __builtin_amdgcn_wmma_f32_16x16x32_f16(false, a_cos1, false, b1,
                                                   (short)0, c_cos, false, false);
    c_sin = __builtin_amdgcn_wmma_f32_16x16x32_f16(false, a_sin0, false, b0,
                                                   (short)0, c_sin, false, false);
    c_sin = __builtin_amdgcn_wmma_f32_16x16x32_f16(false, a_sin1, false, b1,
                                                   (short)0, c_sin, false, false);

    // ---------------- Gather embedding, scale, stream out ------------------
    // C/D layout: VGPR r -> (M = r + 8*half, N = ml)
    const size_t imagOff = (size_t)NBATCH * S_LEN * DMODEL;
    #pragma unroll
    for (int b = 0; b < NBATCH; ++b) {
        #pragma unroll
        for (int r = 0; r < 8; ++r) {
            const int s   = s0 + r + half * 8;
            const int tok = tokens[b * S_LEN + s];
            const float e = emb[(size_t)tok * DMODEL + dcol];
            const size_t o = ((size_t)(b * S_LEN + s)) * DMODEL + dcol;
            __builtin_nontemporal_store(c_cos[r] * e, out + o);            // TH=NT
            __builtin_nontemporal_store(c_sin[r] * e, out + imagOff + o);  // TH=NT
        }
    }
}

extern "C" void kernel_launch(void* const* d_in, const int* in_sizes, int n_in,
                              void* d_out, int out_size, void* d_ws, size_t ws_size,
                              hipStream_t stream)
{
    const int*   tokens = (const int*)d_in[0];
    const int*   t_ptr  = (const int*)d_in[1];
    const float* emb    = (const float*)d_in[2];
    const float* amp    = (const float*)d_in[3];
    const float* freq   = (const float*)d_in[4];
    float*       out    = (float*)d_out;

    dim3 grid(DMODEL / 128, S_LEN / 16);   // (8, 256) blocks
    wave_encoder_fused<<<grid, 256, 0, stream>>>(tokens, t_ptr, emb, amp, freq, out);
}
